// SVDHead_16303695856095
// MI455X (gfx1250) — matrix operand order repaired and checked by reading, not compiled, not run on hardware
//
#include <hip/hip_runtime.h>

// ---------------------------------------------------------------------------
// SVDHead for MI455X (gfx1250, wave32, WMMA).
// Heavy GEMMs (se.se^T, se.te^T) run on v_wmma_f32_16x16x32_bf16 using a
// bf16 hi+lo split (3 WMMAs per 32-K step) for ~fp32 accuracy.
// Per-wave 16x64 output tile: A fragment reused across 4 B tiles
// (12 WMMAs per K-step per wave, ~1.7 b128 loads per WMMA).
// ---------------------------------------------------------------------------

constexpr int NB = 8;     // batch
constexpr int NP = 2048;  // points
constexpr int ND = 512;   // embedding dim
constexpr float F_EPS = 1e-8f;
constexpr float F_BN_EPS = 1e-5f;

typedef __bf16 v16bf __attribute__((ext_vector_type(16)));
typedef float v8f __attribute__((ext_vector_type(8)));

union FragU { unsigned int u[8]; v16bf v; };

__device__ __forceinline__ v8f wmma_bf16(v16bf a, v16bf b, v8f c) {
  return __builtin_amdgcn_wmma_f32_16x16x32_bf16(false, a, false, b,
                                                 (short)0, c, false, false);
}

// A fragment (16x32 bf16, M x K). Layout: lanes0-15: M=lane, K=0..15 in
// v0..7 (sequential pairs); lanes16-31: same M, K=16..31.
__device__ __forceinline__ void load_fragA(const unsigned short* hi,
                                           const unsigned short* lo,
                                           size_t rowbase, int k0, int lane,
                                           v16bf* fh, v16bf* fl) {
  int m = lane & 15;
  int half8 = (lane >> 4) << 4;
  size_t base = rowbase + (size_t)m * ND + (size_t)(k0 + half8);
  const unsigned int* ph = reinterpret_cast<const unsigned int*>(hi) + (base >> 1);
  const unsigned int* pl = reinterpret_cast<const unsigned int*>(lo) + (base >> 1);
  FragU A, L;
#pragma unroll
  for (int p = 0; p < 8; ++p) {
    int o = p;
    A.u[p] = ph[o];
    L.u[p] = pl[o];
  }
  *fh = A.v;
  *fl = L.v;
}

// B fragment (32x16 bf16, K x N). Layout used here (matches A): lanes0-15 hold K=0..15 (v0..7
// sequential pairs) of column N=lane; lanes16-31 hold K=16..31.
// Column n of B == row n of the row-major operand (B = op^T).
__device__ __forceinline__ void load_fragB(const unsigned short* hi,
                                           const unsigned short* lo,
                                           size_t rowbase, int k0, int lane,
                                           v16bf* fh, v16bf* fl) {
  int n = lane & 15;
  int kb = (lane >> 4) << 4;
  size_t base = rowbase + (size_t)n * ND + (size_t)(k0 + kb);
  const unsigned int* ph = reinterpret_cast<const unsigned int*>(hi) + (base >> 1);
  const unsigned int* pl = reinterpret_cast<const unsigned int*>(lo) + (base >> 1);
  FragU A, L;
#pragma unroll
  for (int p = 0; p < 8; ++p) {
    A.u[p] = ph[p];
    L.u[p] = pl[p];
  }
  *fh = A.v;
  *fl = L.v;
}

__device__ __forceinline__ float blockSum(float v, float* red) {
  int t = threadIdx.x;
  red[t] = v;
  __syncthreads();
  for (int s = 128; s > 0; s >>= 1) {
    if (t < s) red[t] += red[t + s];
    __syncthreads();
  }
  float r = red[0];
  __syncthreads();
  return r;
}

__device__ __forceinline__ float blockMax(float v, float* red) {
  int t = threadIdx.x;
  red[t] = v;
  __syncthreads();
  for (int s = 128; s > 0; s >>= 1) {
    if (t < s) red[t] = fmaxf(red[t], red[t + s]);
    __syncthreads();
  }
  float r = red[0];
  __syncthreads();
  return r;
}

// ------------------- transpose (B,D,N)->(B,N,D) + bf16 hi/lo split ----------
__global__ __launch_bounds__(256) void k_trsplit(
    const float* __restrict__ se, const float* __restrict__ te,
    unsigned short* __restrict__ seh, unsigned short* __restrict__ sel,
    unsigned short* __restrict__ teh, unsigned short* __restrict__ tel) {
  int z = blockIdx.z;  // b*2 + which
  int b = z >> 1, e = z & 1;
  const float* src = (e ? te : se) + (size_t)b * ND * NP;
  unsigned short* oh = (e ? teh : seh) + (size_t)b * NP * ND;
  unsigned short* ol = (e ? tel : sel) + (size_t)b * NP * ND;
  __shared__ float tile[32][33];
  int n0 = blockIdx.x * 32, d0 = blockIdx.y * 32;
  int tx = threadIdx.x & 31, ty = threadIdx.x >> 5;  // 32 x 8
#pragma unroll
  for (int i = 0; i < 32; i += 8)
    tile[ty + i][tx] = src[(size_t)(d0 + ty + i) * NP + n0 + tx];
  __syncthreads();
#pragma unroll
  for (int i = 0; i < 32; i += 8) {
    float x = tile[tx][ty + i];
    unsigned int u = __float_as_uint(x);
    unsigned int hb = (u + 0x7FFFu + ((u >> 16) & 1u)) >> 16;  // RNE bf16
    float hf = __uint_as_float(hb << 16);
    float lf = x - hf;
    unsigned int ul = __float_as_uint(lf);
    unsigned int lb = (ul + 0x7FFFu + ((ul >> 16) & 1u)) >> 16;
    size_t o = (size_t)(n0 + ty + i) * ND + d0 + tx;
    oh[o] = (unsigned short)hb;
    ol[o] = (unsigned short)lb;
  }
}

// ------------------------------- squared norms ------------------------------
__global__ __launch_bounds__(256) void k_sqnorm(const float* __restrict__ se,
                                                const float* __restrict__ te,
                                                float* __restrict__ sqse,
                                                float* __restrict__ sqte) {
  int b = blockIdx.y, e = blockIdx.z;
  int n = blockIdx.x * 256 + threadIdx.x;
  const float* src = (e ? te : se) + (size_t)b * ND * NP;
  float s = 0.0f;
  for (int d = 0; d < ND; ++d) {
    float x = src[(size_t)d * NP + n];
    s += x * x;
  }
  (e ? sqte : sqse)[b * NP + n] = s;
}

// ------------------------------- feat row means -----------------------------
__global__ __launch_bounds__(256) void k_feat(const float* __restrict__ se,
                                              const float* __restrict__ te,
                                              float* __restrict__ feat) {
  int d = blockIdx.x, b = blockIdx.y;
  __shared__ float red[256];
  float s = 0.0f;
  const float* ps = se + ((size_t)b * ND + d) * NP;
  const float* pt = te + ((size_t)b * ND + d) * NP;
  for (int m = threadIdx.x; m < NP; m += 256) s += ps[m] - pt[m];
  float tot = blockSum(s, red);
  if (threadIdx.x == 0) feat[b * ND + d] = fabsf(tot / (float)NP);
}

// ------------------------------- init min-d2 --------------------------------
__global__ __launch_bounds__(256) void k_init_min(unsigned int* p) {
  p[blockIdx.x * 256 + threadIdx.x] = 0x7F800000u;  // +inf
}

// ------------------- self GEMM + fused row-min (alpha) ----------------------
// Per-wave 16x64 tile: i = blockIdx.y*128 + wave*16, j = blockIdx.x*64.
__global__ __launch_bounds__(256) void k_gemm_self(
    const unsigned short* __restrict__ seh, const unsigned short* __restrict__ sel,
    const float* __restrict__ sq, unsigned int* __restrict__ mind2) {
  int b = blockIdx.z;
  int lane = threadIdx.x & 31, w = threadIdx.x >> 5;
  int i0 = blockIdx.y * 128 + w * 16;
  int j0 = blockIdx.x * 64;
  size_t arow = ((size_t)b * NP + i0) * ND;
  size_t brow = ((size_t)b * NP + j0) * ND;
  v8f c[4];
#pragma unroll
  for (int jj = 0; jj < 4; ++jj)
    c[jj] = v8f{0.f, 0.f, 0.f, 0.f, 0.f, 0.f, 0.f, 0.f};
  for (int k0 = 0; k0 < ND; k0 += 32) {
    v16bf ah, al;
    load_fragA(seh, sel, arow, k0, lane, &ah, &al);
#pragma unroll
    for (int jj = 0; jj < 4; ++jj) {
      v16bf bh, bl;
      load_fragB(seh, sel, brow + (size_t)jj * 16 * ND, k0, lane, &bh, &bl);
      c[jj] = wmma_bf16(ah, bh, c[jj]);
      c[jj] = wmma_bf16(ah, bl, c[jj]);
      c[jj] = wmma_bf16(al, bh, c[jj]);
    }
  }
  int m = lane & 15, hb = lane >> 4;
  float sqj[4];
#pragma unroll
  for (int jj = 0; jj < 4; ++jj) sqj[jj] = sq[b * NP + j0 + jj * 16 + m];
#pragma unroll
  for (int r = 0; r < 8; ++r) {
    int ig = i0 + r + 8 * hb;
    float sqi = sq[b * NP + ig];
    float dmin = 3.0e38f;
#pragma unroll
    for (int jj = 0; jj < 4; ++jj) {
      int jg = j0 + jj * 16 + m;
      float d2 = sqi + sqj[jj] - 2.0f * c[jj][r];
      d2 = fmaxf(d2, 0.0f);
      if (ig == jg) d2 = 3.0e38f;  // exclude self
      dmin = fminf(dmin, d2);
    }
#pragma unroll
    for (int msk = 8; msk >= 1; msk >>= 1)
      dmin = fminf(dmin, __shfl_xor(dmin, msk, 32));
    if (m == 0) atomicMin(&mind2[b * NP + ig], __float_as_uint(dmin));
  }
}

// ------------------------------- alpha -> temp ------------------------------
__global__ __launch_bounds__(256) void k_alpha(const unsigned int* __restrict__ mind2,
                                               float* __restrict__ temp) {
  int b = blockIdx.x;
  __shared__ float red[256];
  float s = 0.0f;
  for (int n = threadIdx.x; n < NP; n += 256)
    s += sqrtf(__uint_as_float(mind2[b * NP + n]));
  float tot = blockSum(s, red);
  if (threadIdx.x == 0) temp[b] = tot / (float)NP;  // ALPHA_FACTOR = 1
}

// ---------------------- cross GEMM -> d_emb (B,N,N) -------------------------
__global__ __launch_bounds__(256) void k_gemm_cross(
    const unsigned short* __restrict__ seh, const unsigned short* __restrict__ sel,
    const unsigned short* __restrict__ teh, const unsigned short* __restrict__ tel,
    const float* __restrict__ sqse, const float* __restrict__ sqte,
    float* __restrict__ demb) {
  int b = blockIdx.z;
  int lane = threadIdx.x & 31, w = threadIdx.x >> 5;
  int i0 = blockIdx.y * 128 + w * 16;
  int j0 = blockIdx.x * 64;
  size_t arow = ((size_t)b * NP + i0) * ND;
  size_t brow = ((size_t)b * NP + j0) * ND;
  v8f c[4];
#pragma unroll
  for (int jj = 0; jj < 4; ++jj)
    c[jj] = v8f{0.f, 0.f, 0.f, 0.f, 0.f, 0.f, 0.f, 0.f};
  for (int k0 = 0; k0 < ND; k0 += 32) {
    v16bf ah, al;
    load_fragA(seh, sel, arow, k0, lane, &ah, &al);
#pragma unroll
    for (int jj = 0; jj < 4; ++jj) {
      v16bf bh, bl;
      load_fragB(teh, tel, brow + (size_t)jj * 16 * ND, k0, lane, &bh, &bl);
      c[jj] = wmma_bf16(ah, bh, c[jj]);
      c[jj] = wmma_bf16(ah, bl, c[jj]);
      c[jj] = wmma_bf16(al, bh, c[jj]);
    }
  }
  int m = lane & 15, hb = lane >> 4;
  float sqj[4];
#pragma unroll
  for (int jj = 0; jj < 4; ++jj) sqj[jj] = sqte[b * NP + j0 + jj * 16 + m];
#pragma unroll
  for (int r = 0; r < 8; ++r) {
    int ig = i0 + r + 8 * hb;
    float sqi = sqse[b * NP + ig];
    float* orow = demb + ((size_t)b * NP + ig) * NP;
#pragma unroll
    for (int jj = 0; jj < 4; ++jj) {
      float d2 = sqi + sqj[jj] - 2.0f * c[jj][r];
      orow[j0 + jj * 16 + m] = sqrtf(fmaxf(d2, 0.0f));
    }
  }
}

// ------------------------------- T-net MLP ----------------------------------
__global__ __launch_bounds__(128) void k_mlp(
    const float* __restrict__ feat,
    const float* __restrict__ W1, const float* __restrict__ b1,
    const float* __restrict__ g1, const float* __restrict__ be1,
    const float* __restrict__ W2, const float* __restrict__ b2,
    const float* __restrict__ g2, const float* __restrict__ be2,
    const float* __restrict__ W3, const float* __restrict__ b3,
    const float* __restrict__ g3, const float* __restrict__ be3,
    const float* __restrict__ W4, const float* __restrict__ b4,
    const int* __restrict__ iter_num, float* __restrict__ Tv) {
  __shared__ float X[NB * ND];   // 16 KB
  __shared__ float H[NB * 128];  // 4 KB
  int j = threadIdx.x;
  for (int i = j; i < NB * ND; i += 128) X[i] = feat[i];
  __syncthreads();
  float h[NB];
  // layer 1: 512 -> 128, BN, relu
  for (int r = 0; r < NB; ++r) {
    float s = b1[j];
    for (int k = 0; k < ND; ++k) s += X[r * ND + k] * W1[k * 128 + j];
    h[r] = s;
  }
  {
    float mu = 0.f;
    for (int r = 0; r < NB; ++r) mu += h[r];
    mu *= (1.0f / NB);
    float var = 0.f;
    for (int r = 0; r < NB; ++r) { float d = h[r] - mu; var += d * d; }
    var *= (1.0f / NB);
    float sc = g1[j] * rsqrtf(var + F_BN_EPS);
    for (int r = 0; r < NB; ++r)
      H[r * 128 + j] = fmaxf(0.f, (h[r] - mu) * sc + be1[j]);
  }
  __syncthreads();
  // layer 2: 128 -> 128 (H -> X)
  for (int r = 0; r < NB; ++r) {
    float s = b2[j];
    for (int k = 0; k < 128; ++k) s += H[r * 128 + k] * W2[k * 128 + j];
    h[r] = s;
  }
  {
    float mu = 0.f;
    for (int r = 0; r < NB; ++r) mu += h[r];
    mu *= (1.0f / NB);
    float var = 0.f;
    for (int r = 0; r < NB; ++r) { float d = h[r] - mu; var += d * d; }
    var *= (1.0f / NB);
    float sc = g2[j] * rsqrtf(var + F_BN_EPS);
    for (int r = 0; r < NB; ++r)
      h[r] = fmaxf(0.f, (h[r] - mu) * sc + be2[j]);
  }
  __syncthreads();
  for (int r = 0; r < NB; ++r) X[r * 128 + j] = h[r];
  __syncthreads();
  // layer 3: 128 -> 128 (X -> H)
  for (int r = 0; r < NB; ++r) {
    float s = b3[j];
    for (int k = 0; k < 128; ++k) s += X[r * 128 + k] * W3[k * 128 + j];
    h[r] = s;
  }
  {
    float mu = 0.f;
    for (int r = 0; r < NB; ++r) mu += h[r];
    mu *= (1.0f / NB);
    float var = 0.f;
    for (int r = 0; r < NB; ++r) { float d = h[r] - mu; var += d * d; }
    var *= (1.0f / NB);
    float sc = g3[j] * rsqrtf(var + F_BN_EPS);
    for (int r = 0; r < NB; ++r)
      H[r * 128 + j] = fmaxf(0.f, (h[r] - mu) * sc + be3[j]);
  }
  __syncthreads();
  // layer 4: 128 -> 1, clip, /2^(iter-1)
  if (j < NB) {
    float s = b4[0];
    for (int k = 0; k < 128; ++k) s += H[j * 128 + k] * W4[k];
    s = fminf(fmaxf(fmaxf(s, 0.f), 0.01f), 100.0f);
    Tv[j] = s * exp2f((float)(1 - iter_num[0]));
  }
}

// ---------------------- row softmax stats over neg = -d/t -------------------
__global__ __launch_bounds__(256) void k_rowstats(const float* __restrict__ demb,
                                                  const float* __restrict__ temp,
                                                  float* __restrict__ rmax,
                                                  float* __restrict__ rsum) {
  int n = blockIdx.x, b = blockIdx.y;
  const float* row = demb + ((size_t)b * NP + n) * NP;
  float it = 1.0f / temp[b];
  __shared__ float red[256];
  float mx = -3.0e38f;
  for (int m = threadIdx.x; m < NP; m += 256) mx = fmaxf(mx, -row[m] * it);
  float bmx = blockMax(mx, red);
  float s = 0.0f;
  for (int m = threadIdx.x; m < NP; m += 256) s += __expf(-row[m] * it - bmx);
  float bs = blockSum(s, red);
  if (threadIdx.x == 0) {
    rmax[b * NP + n] = bmx;
    rsum[b * NP + n] = bs;
  }
}

// ------------------- column softmax stats (online, coalesced) ---------------
__global__ __launch_bounds__(256) void k_colstats(const float* __restrict__ demb,
                                                  const float* __restrict__ temp,
                                                  float* __restrict__ cmax,
                                                  float* __restrict__ csum) {
  int b = blockIdx.y;
  int m = blockIdx.x * 256 + threadIdx.x;
  float it = 1.0f / temp[b];
  float M = -3.0e38f, S = 0.0f;
  const float* base = demb + (size_t)b * NP * NP + m;
  for (int n = 0; n < NP; ++n) {
    float x = -base[(size_t)n * NP] * it;
    if (x > M) {
      S = S * __expf(M - x) + 1.0f;
      M = x;
    } else {
      S += __expf(x - M);
    }
  }
  cmax[b * NP + m] = M;
  csum[b * NP + m] = S;
}

// --- fused pass: scores, row-sum, gamma (with euclid), src_corr output ------
__global__ __launch_bounds__(256) void k_pass3(
    const float* __restrict__ demb, const float* __restrict__ temp,
    const float* __restrict__ rmax, const float* __restrict__ rsum,
    const float* __restrict__ cmax, const float* __restrict__ csum,
    const float* __restrict__ srcp, const float* __restrict__ tgtp,
    const float* __restrict__ Tv, float* __restrict__ gamma,
    float* __restrict__ corr_out) {
  int n = blockIdx.x, b = blockIdx.y;
  const float* row = demb + ((size_t)b * NP + n) * NP;
  float it = 1.0f / temp[b];
  float invT = 1.0f / Tv[b];
  float rm = rmax[b * NP + n];
  float rinv = 1.0f / rsum[b * NP + n];
  float sx = srcp[((size_t)b * 3 + 0) * NP + n];
  float sy = srcp[((size_t)b * 3 + 1) * NP + n];
  float sz = srcp[((size_t)b * 3 + 2) * NP + n];
  const float* tx = tgtp + ((size_t)b * 3 + 0) * NP;
  const float* ty = tgtp + ((size_t)b * 3 + 1) * NP;
  const float* tz = tgtp + ((size_t)b * 3 + 2) * NP;
  float sumS = 0.f, g = 0.f, c0 = 0.f, c1 = 0.f, c2 = 0.f;
  for (int m = threadIdx.x; m < NP; m += 256) {
    float d = row[m];
    // scores = softmax_col * softmax_row = exp(2*neg - rmax - cmax)/(rs*cs)
    float s = __expf(-2.0f * d * it - rm - cmax[b * NP + m]) * rinv /
              csum[b * NP + m];
    sumS += s;
    float x0 = tx[m], x1 = ty[m], x2 = tz[m];
    float dx = sx - x0, dy = sy - x1, dz = sz - x2;
    float euc = sqrtf(dx * dx + dy * dy + dz * dz);
    g += s * __expf(-euc * invT);
    c0 += s * x0;
    c1 += s * x1;
    c2 += s * x2;
  }
  __shared__ float red[256];
  float tS = blockSum(sumS, red);
  float tg = blockSum(g, red);
  float t0 = blockSum(c0, red);
  float t1 = blockSum(c1, red);
  float t2 = blockSum(c2, red);
  if (threadIdx.x == 0) {
    float inv = 1.0f / (tS + F_EPS);
    corr_out[((size_t)b * 3 + 0) * NP + n] = t0 * inv;
    corr_out[((size_t)b * 3 + 1) * NP + n] = t1 * inv;
    corr_out[((size_t)b * 3 + 2) * NP + n] = t2 * inv;
    gamma[b * NP + n] = tg;
  }
}

// ------------------------------ 3x3 helpers ---------------------------------
__device__ __forceinline__ float det3(const float M[3][3]) {
  return M[0][0] * (M[1][1] * M[2][2] - M[1][2] * M[2][1]) -
         M[0][1] * (M[1][0] * M[2][2] - M[1][2] * M[2][0]) +
         M[0][2] * (M[1][0] * M[2][1] - M[1][1] * M[2][0]);
}

__device__ void jrot(float S[3][3], float V[3][3], int p, int q) {
  float apq = S[p][q];
  if (fabsf(apq) < 1e-20f) return;
  float tau = (S[q][q] - S[p][p]) / (2.0f * apq);
  float t = (tau >= 0.0f ? 1.0f : -1.0f) / (fabsf(tau) + sqrtf(1.0f + tau * tau));
  float cc = rsqrtf(1.0f + t * t);
  float ss = t * cc;
  for (int k = 0; k < 3; ++k) {
    float a = S[k][p], bb = S[k][q];
    S[k][p] = cc * a - ss * bb;
    S[k][q] = ss * a + cc * bb;
  }
  for (int k = 0; k < 3; ++k) {
    float a = S[p][k], bb = S[q][k];
    S[p][k] = cc * a - ss * bb;
    S[q][k] = ss * a + cc * bb;
  }
  for (int k = 0; k < 3; ++k) {
    float a = V[k][p], bb = V[k][q];
    V[k][p] = cc * a - ss * bb;
    V[k][q] = ss * a + cc * bb;
  }
}

// ----------------- per-batch moments, H, SVD, R & t -------------------------
__global__ __launch_bounds__(256) void k_finalize(
    const float* __restrict__ gamma, const float* __restrict__ srcp,
    const float* __restrict__ corr, float* __restrict__ out) {
  int b = blockIdx.x;
  float acc[16];
#pragma unroll
  for (int q = 0; q < 16; ++q) acc[q] = 0.f;
  for (int n = threadIdx.x; n < NP; n += 256) {
    float g = gamma[b * NP + n];
    float sv[3], cv[3];
#pragma unroll
    for (int d = 0; d < 3; ++d) {
      sv[d] = srcp[((size_t)b * 3 + d) * NP + n];
      cv[d] = corr[((size_t)b * 3 + d) * NP + n];
    }
    acc[0] += g;
#pragma unroll
    for (int d = 0; d < 3; ++d) {
      acc[1 + d] += g * sv[d];
      acc[4 + d] += g * cv[d];
#pragma unroll
      for (int e = 0; e < 3; ++e) acc[7 + d * 3 + e] += g * sv[d] * cv[e];
    }
  }
  __shared__ float red[256];
  float tot[16];
  for (int q = 0; q < 16; ++q) tot[q] = blockSum(acc[q], red);
  if (threadIdx.x != 0) return;

  float G0 = tot[0];
  float Av[3] = {tot[1], tot[2], tot[3]};
  float Bw[3] = {tot[4], tot[5], tot[6]};
  float gsum = G0 + F_EPS;
  float sm[3], cm[3];
  for (int d = 0; d < 3; ++d) {
    sm[d] = Av[d] / gsum;
    cm[d] = Bw[d] / gsum;
  }
  float Hm[3][3];
  for (int d = 0; d < 3; ++d)
    for (int e = 0; e < 3; ++e)
      Hm[d][e] = tot[7 + d * 3 + e] - sm[d] * Bw[e] - Av[d] * cm[e] +
                 G0 * sm[d] * cm[e];
  Hm[0][0] += F_EPS * 1.0f;
  Hm[1][1] += F_EPS * 2.0f;
  Hm[2][2] += F_EPS * 3.0f;

  // SVD via Jacobi eigendecomposition of H^T H
  float S[3][3];
  for (int i = 0; i < 3; ++i)
    for (int j = 0; j < 3; ++j)
      S[i][j] = Hm[0][i] * Hm[0][j] + Hm[1][i] * Hm[1][j] + Hm[2][i] * Hm[2][j];
  float V[3][3] = {{1.f, 0.f, 0.f}, {0.f, 1.f, 0.f}, {0.f, 0.f, 1.f}};
  for (int it = 0; it < 15; ++it) {
    jrot(S, V, 0, 1);
    jrot(S, V, 0, 2);
    jrot(S, V, 1, 2);
  }
  float ev[3] = {S[0][0], S[1][1], S[2][2]};
  for (int a = 0; a < 2; ++a)
    for (int c = a + 1; c < 3; ++c)
      if (ev[c] > ev[a]) {
        float tmp = ev[a]; ev[a] = ev[c]; ev[c] = tmp;
        for (int k = 0; k < 3; ++k) {
          float tv2 = V[k][a]; V[k][a] = V[k][c]; V[k][c] = tv2;
        }
      }
  float U[3][3];
  for (int i = 0; i < 3; ++i) {
    float u0 = Hm[0][0] * V[0][i] + Hm[0][1] * V[1][i] + Hm[0][2] * V[2][i];
    float u1 = Hm[1][0] * V[0][i] + Hm[1][1] * V[1][i] + Hm[1][2] * V[2][i];
    float u2 = Hm[2][0] * V[0][i] + Hm[2][1] * V[1][i] + Hm[2][2] * V[2][i];
    float nn = sqrtf(u0 * u0 + u1 * u1 + u2 * u2);
    if (nn > 1e-12f) {
      u0 /= nn; u1 /= nn; u2 /= nn;
    } else if (i == 2) {
      u0 = U[1][0] * U[2][1] - U[2][0] * U[1][1];
      u1 = U[2][0] * U[0][1] - U[0][0] * U[2][1];
      u2 = U[0][0] * U[1][1] - U[1][0] * U[0][1];
      float n2 = sqrtf(u0 * u0 + u1 * u1 + u2 * u2);
      if (n2 > 1e-12f) { u0 /= n2; u1 /= n2; u2 /= n2; }
    }
    U[0][i] = u0; U[1][i] = u1; U[2][i] = u2;
  }
  float Rm[3][3];
  for (int a = 0; a < 3; ++a)
    for (int c = 0; c < 3; ++c)
      Rm[a][c] = V[a][0] * U[c][0] + V[a][1] * U[c][1] + V[a][2] * U[c][2];
  if (det3(Rm) < 0.0f) {
    for (int k = 0; k < 3; ++k) V[k][2] = -V[k][2];
    for (int a = 0; a < 3; ++a)
      for (int c = 0; c < 3; ++c)
        Rm[a][c] = V[a][0] * U[c][0] + V[a][1] * U[c][1] + V[a][2] * U[c][2];
  }
  float tv[3];
  for (int a = 0; a < 3; ++a)
    tv[a] = -(Rm[a][0] * sm[0] + Rm[a][1] * sm[1] + Rm[a][2] * sm[2]) + cm[a];

  for (int a = 0; a < 3; ++a) {
    for (int c = 0; c < 3; ++c) out[b * 9 + a * 3 + c] = Rm[a][c];
    out[NB * 9 + b * 3 + a] = tv[a];
  }
}

// ---------------------------------------------------------------------------
extern "C" void kernel_launch(void* const* d_in, const int* in_sizes, int n_in,
                              void* d_out, int out_size, void* d_ws,
                              size_t ws_size, hipStream_t stream) {
  const float* se = (const float*)d_in[0];   // (B,D,N)
  const float* te = (const float*)d_in[1];   // (B,D,N)
  const float* srcp = (const float*)d_in[2]; // (B,3,N)
  const float* tgtp = (const float*)d_in[3]; // (B,3,N)
  const float* W1 = (const float*)d_in[4];
  const float* b1 = (const float*)d_in[5];
  const float* g1 = (const float*)d_in[6];
  const float* be1 = (const float*)d_in[7];
  const float* W2 = (const float*)d_in[8];
  const float* b2 = (const float*)d_in[9];
  const float* g2 = (const float*)d_in[10];
  const float* be2 = (const float*)d_in[11];
  const float* W3 = (const float*)d_in[12];
  const float* b3 = (const float*)d_in[13];
  const float* g3 = (const float*)d_in[14];
  const float* be3 = (const float*)d_in[15];
  const float* W4 = (const float*)d_in[16];
  const float* b4 = (const float*)d_in[17];
  const int* iter_num = (const int*)d_in[18];

  float* out = (float*)d_out;
  float* corr_out = out + NB * 9 + NB * 3;  // src_corr (B,3,N)

  // workspace layout
  const size_t SZE = (size_t)NB * NP * ND * sizeof(unsigned short);  // 16 MB
  const size_t SZNN = (size_t)NB * NP * NP * sizeof(float);          // 128 MB
  const size_t SZBN = (size_t)NB * NP * sizeof(float);               // 64 KB
  char* w = (char*)d_ws;
  unsigned short* seh = (unsigned short*)(w + 0 * SZE);
  unsigned short* sel = (unsigned short*)(w + 1 * SZE);
  unsigned short* teh = (unsigned short*)(w + 2 * SZE);
  unsigned short* tel = (unsigned short*)(w + 3 * SZE);
  float* demb = (float*)(w + 4 * SZE);
  char* w2 = w + 4 * SZE + SZNN;
  float* sqse = (float*)(w2 + 0 * SZBN);
  float* sqte = (float*)(w2 + 1 * SZBN);
  unsigned int* mind2 = (unsigned int*)(w2 + 2 * SZBN);
  float* rmax = (float*)(w2 + 3 * SZBN);
  float* rsum = (float*)(w2 + 4 * SZBN);
  float* cmax = (float*)(w2 + 5 * SZBN);
  float* csum = (float*)(w2 + 6 * SZBN);
  float* gamma = (float*)(w2 + 7 * SZBN);
  char* w3 = w2 + 8 * SZBN;
  float* temp = (float*)(w3);
  float* Tv = (float*)(w3 + 256);
  float* feat = (float*)(w3 + 512);  // B*D floats

  k_trsplit<<<dim3(NP / 32, ND / 32, 2 * NB), 256, 0, stream>>>(
      se, te, seh, sel, teh, tel);
  k_sqnorm<<<dim3(NP / 256, NB, 2), 256, 0, stream>>>(se, te, sqse, sqte);
  k_feat<<<dim3(ND, NB), 256, 0, stream>>>(se, te, feat);
  k_init_min<<<NB * NP / 256, 256, 0, stream>>>(mind2);
  k_gemm_self<<<dim3(NP / 64, NP / 128, NB), 256, 0, stream>>>(seh, sel, sqse,
                                                               mind2);
  k_alpha<<<NB, 256, 0, stream>>>(mind2, temp);
  k_gemm_cross<<<dim3(NP / 64, NP / 128, NB), 256, 0, stream>>>(
      seh, sel, teh, tel, sqse, sqte, demb);
  k_mlp<<<1, 128, 0, stream>>>(feat, W1, b1, g1, be1, W2, b2, g2, be2, W3, b3,
                               g3, be3, W4, b4, iter_num, Tv);
  k_rowstats<<<dim3(NP, NB), 256, 0, stream>>>(demb, temp, rmax, rsum);
  k_colstats<<<dim3(NP / 256, NB), 256, 0, stream>>>(demb, temp, cmax, csum);
  k_pass3<<<dim3(NP, NB), 256, 0, stream>>>(demb, temp, rmax, rsum, cmax, csum,
                                            srcp, tgtp, Tv, gamma, corr_out);
  k_finalize<<<NB, 256, 0, stream>>>(gamma, srcp, corr_out, out);
}
